// WindowAttention_14336600834053
// MI455X (gfx1250) — matrix operand order repaired and checked by reading, not compiled, not run on hardware
//
#include <hip/hip_runtime.h>
#include <hip/hip_bf16.h>

typedef __attribute__((ext_vector_type(16))) _Float16 v16h;
typedef __attribute__((ext_vector_type(8)))  float    v8f;

#define WS        8
#define NTOK      64
#define DIM       256
#define HEADS     8
#define NWIN      4096

#define XSTRIDE   264          // 256 + 8 pad (f16 units)
#define VSTRIDE   72           // 64 + 8 pad
#define PSTRIDE   72

// LDS layout: table f32 first, then f16 arrays.
#define TB_BYTES   7200
#define F16_BASE   (TB_BYTES / 2)
#define XH_OFF     (F16_BASE)
#define QH_OFF     (XH_OFF + 64 * XSTRIDE)
#define KH_OFF     (QH_OFF + 64 * XSTRIDE)
#define OH_OFF     (KH_OFF + 64 * XSTRIDE)
#define VT_OFF     (OH_OFF + 64 * XSTRIDE)
#define P_OFF      (VT_OFF + 256 * VSTRIDE)
#define SMEM_F16   (P_OFF + 8 * 64 * PSTRIDE)
#define SMEM_BYTES (SMEM_F16 * 2)                    // = 252960 < 320KB

union FragU { v16h h; unsigned u[8]; };

// A-matrix fragment (16 x 32 K-window at k0), base row-major [row][stride].
// lane = M; VGPR v<4: K=2v(+8 for hi lanes); v>=4: K=16+2(v-4)(+8)
__device__ __forceinline__ v16h load_fragA(const _Float16* base, int row0, int stride,
                                           int k0, int lane) {
  FragU f;
  const _Float16* rp = base + (row0 + (lane & 15)) * stride + k0 + ((lane >> 4) << 3);
#pragma unroll
  for (int v = 0; v < 4; ++v) f.u[v]     = *(const unsigned*)(rp + 2 * v);
#pragma unroll
  for (int v = 0; v < 4; ++v) f.u[4 + v] = *(const unsigned*)(rp + 16 + 2 * v);
  return f.h;
}

// B-matrix fragment (32 x 16). base holds B^T row-major: base[n][k].
// lane = N; VGPR v<4: K=2v(+8 for hi lanes); v>=4: K=16+2(v-4)(+8)
__device__ __forceinline__ v16h load_fragB(const _Float16* base, int col0, int stride,
                                           int k0, int lane) {
  FragU f;
  const _Float16* rp = base + (col0 + (lane & 15)) * stride + k0 + ((lane >> 4) << 3);
#pragma unroll
  for (int v = 0; v < 8; ++v) f.u[v] = *(const unsigned*)(rp + 2 * v + ((v >> 2) << 3));
  return f.h;
}

__device__ __forceinline__ v8f wmma16(v16h a, v16h b, v8f c) {
  return __builtin_amdgcn_wmma_f32_16x16x32_f16(false, a, false, b, (short)0, c, false, false);
}

// Store 16x16 f32 D tile as f16, row-major dest; D layout: VGPR r -> M=r+8*(lane/16)
__device__ __forceinline__ void store_tile_f16(_Float16* base, int row0, int col0, int stride,
                                               int lane, v8f acc, float scl, float badd) {
  int col = col0 + (lane & 15);
  int rb  = row0 + ((lane >> 4) << 3);
#pragma unroll
  for (int r = 0; r < 8; ++r)
    base[(rb + r) * stride + col] = (_Float16)(acc[r] * scl + badd);
}

// Transposed store: dest[col][row] (V^T) -> contiguous 8 f16 per lane (b128 store)
__device__ __forceinline__ void store_tileT_f16(_Float16* base, int row0, int col0, int stride,
                                                int lane, v8f acc, float badd) {
  int col = col0 + (lane & 15);
  int rb  = row0 + ((lane >> 4) << 3);
#pragma unroll
  for (int r = 0; r < 8; ++r)
    base[col * stride + rb + r] = (_Float16)(acc[r] + badd);
}

// ---------------------------------------------------------------------------
// Weight pre-pass: f32 [in][out] -> f16 transposed [out][in], 4 matrices packed:
//   wT[0]=wq^T, wT[1]=wk^T (wkv cols 0..255), wT[2]=wv^T (cols 256..511), wT[3]=wp^T
// ---------------------------------------------------------------------------
__global__ void wa_convert_weights(const float* __restrict__ wq,
                                   const float* __restrict__ wkv,
                                   const float* __restrict__ wp,
                                   _Float16* __restrict__ wT) {
  int i = blockIdx.x * blockDim.x + threadIdx.x;
  if (i >= 65536) return;
  int n = i >> 8;        // out feature
  int k = i & 255;       // in feature
  wT[i]          = (_Float16)wq [k * 256 + n];
  wT[65536 + i]  = (_Float16)wkv[k * 512 + n];
  wT[131072 + i] = (_Float16)wkv[k * 512 + 256 + n];
  wT[196608 + i] = (_Float16)wp [k * 256 + n];
}

// ---------------------------------------------------------------------------
// Fused window attention: one workgroup (8 waves) per window; wave h = head h.
// ---------------------------------------------------------------------------
__global__ __launch_bounds__(256) void wa_attn_kernel(const float* __restrict__ x,
                                                      const float* __restrict__ bq,
                                                      const float* __restrict__ bkv,
                                                      const float* __restrict__ table,
                                                      const float* __restrict__ bp,
                                                      const _Float16* __restrict__ wT,
                                                      float* __restrict__ out) {
  extern __shared__ char smem_raw[];
  float*    tb = (float*)smem_raw;
  _Float16* sh = (_Float16*)smem_raw;
  _Float16* xh = sh + XH_OFF;
  _Float16* qh = sh + QH_OFF;
  _Float16* kh = sh + KH_OFF;
  _Float16* oh = sh + OH_OFF;
  _Float16* vt = sh + VT_OFF;
  _Float16* pb = sh + P_OFF;

  const int tid  = threadIdx.x;
  const int lane = tid & 31;
  const int wave = __builtin_amdgcn_readfirstlane(tid >> 5);  // force SGPR: scalar branches
  const int win  = blockIdx.x;

  // --- stage: bias table -> LDS ---
  for (int i = tid; i < 225 * HEADS; i += 256) tb[i] = table[i];

  // --- stage: x (64x256 f32) -> LDS f16 ---
  const float* xg = x + (size_t)win * NTOK * DIM;
  for (int i = tid * 4; i < NTOK * DIM; i += 256 * 4) {
    float4 v = *(const float4*)(xg + i);
    int r = i >> 8, c = i & 255;
    _Float16* d = xh + r * XSTRIDE + c;
    d[0] = (_Float16)v.x; d[1] = (_Float16)v.y;
    d[2] = (_Float16)v.z; d[3] = (_Float16)v.w;
  }
  __syncthreads();

  // --- phase 1: QKV projection. 48 N-tiles across 8 waves (6 each), K=256. ---
  const float scale = 0.1767766952966369f;  // 32^-0.5
  for (int t = 0; t < 6; ++t) {
    int ntile = wave * 6 + t;            // 0..47, wave-uniform (SGPR)
    int j0    = ntile << 4;
    int sel   = j0 >> 8;                 // 0=q 1=k 2=v  (scalar)
    int col0  = j0 & 255;
    const _Float16* w = wT + (size_t)sel * 65536;
    // hoist all 8 K-step weight fragments: reused across 4 M-tiles (4x less L2 traffic)
    v16h bf[8];
#pragma unroll
    for (int ks = 0; ks < 8; ++ks) bf[ks] = load_fragB(w, col0, 256, ks * 32, lane);
    for (int mt = 0; mt < 4; ++mt) {
      v8f acc = {};
#pragma unroll
      for (int ks = 0; ks < 8; ++ks) {
        v16h a = load_fragA(xh, mt * 16, XSTRIDE, ks * 32, lane);
        acc = wmma16(a, bf[ks], acc);
      }
      if (sel == 0) {
        float bb = bq[col0 + (lane & 15)];
        store_tile_f16(qh, mt * 16, col0, XSTRIDE, lane, acc, scale, bb * scale);
      } else if (sel == 1) {
        float bb = bkv[col0 + (lane & 15)];
        store_tile_f16(kh, mt * 16, col0, XSTRIDE, lane, acc, 1.0f, bb);
      } else {
        float bb = bkv[256 + col0 + (lane & 15)];
        store_tileT_f16(vt, mt * 16, col0, VSTRIDE, lane, acc, bb);  // V^T [feat][tok]
      }
    }
  }
  __syncthreads();

  // --- phase 2: per-head attention; wave h owns head h ---
  const int h = wave;
  _Float16* ph = pb + h * 64 * PSTRIDE;
  const int N0 = lane & 15;

  // hoist the 4 K-row fragments (B of Q.K^T): reused across all 4 M-tiles
  v16h kf[4];
#pragma unroll
  for (int nt = 0; nt < 4; ++nt)
    kf[nt] = load_fragA(kh, nt * 16, XSTRIDE, h * 32, lane);

  for (int mt = 0; mt < 4; ++mt) {
    v16h a = load_fragA(qh, mt * 16, XSTRIDE, h * 32, lane);
    v8f s[4];
#pragma unroll
    for (int nt = 0; nt < 4; ++nt) {
      v8f z = {};
      s[nt] = wmma16(a, kf[nt], z);
    }
    // relative-position bias (analytic index, LDS table gather)
    int Mb = mt * 16 + ((lane >> 4) << 3);
#pragma unroll
    for (int r = 0; r < 8; ++r) {
      int M = Mb + r, iM = M >> 3, jM = M & 7;
#pragma unroll
      for (int nt = 0; nt < 4; ++nt) {
        int N = nt * 16 + N0, iN = N >> 3, jN = N & 7;
        int idx = (iM - iN + WS - 1) * (2 * WS - 1) + (jM - jN + WS - 1);
        s[nt][r] += tb[idx * HEADS + h];
      }
    }
    // softmax per row: a row lives in one lane-half across 4 tiles -> xor masks 1..8
#pragma unroll
    for (int r = 0; r < 8; ++r) {
      float a0 = s[0][r], a1 = s[1][r], a2 = s[2][r], a3 = s[3][r];
      float m = fmaxf(fmaxf(a0, a1), fmaxf(a2, a3));
#pragma unroll
      for (int off = 8; off >= 1; off >>= 1) m = fmaxf(m, __shfl_xor(m, off, 32));
      a0 = __expf(a0 - m); a1 = __expf(a1 - m); a2 = __expf(a2 - m); a3 = __expf(a3 - m);
      float sum = a0 + a1 + a2 + a3;
#pragma unroll
      for (int off = 8; off >= 1; off >>= 1) sum += __shfl_xor(sum, off, 32);
      float inv = 1.0f / sum;
      _Float16* pr = ph + (Mb + r) * PSTRIDE + N0;
      pr[0]  = (_Float16)(a0 * inv);
      pr[16] = (_Float16)(a1 * inv);
      pr[32] = (_Float16)(a2 * inv);
      pr[48] = (_Float16)(a3 * inv);
    }
  }
  // p/vt reads below are wave-local or pre-barrier: same-wave LDS ops are in-order.

  // --- P @ V_h -> oh[:, h*32 : h*32+32]; hoist 4 V^T fragments ---
  v16h vf[2][2];
#pragma unroll
  for (int nt = 0; nt < 2; ++nt)
#pragma unroll
    for (int ks = 0; ks < 2; ++ks)
      vf[nt][ks] = load_fragB(vt + (h * 32) * VSTRIDE, nt * 16, VSTRIDE, ks * 32, lane);

  for (int mt = 0; mt < 4; ++mt) {
    v16h a0 = load_fragA(ph, mt * 16, PSTRIDE, 0,  lane);
    v16h a1 = load_fragA(ph, mt * 16, PSTRIDE, 32, lane);
#pragma unroll
    for (int nt = 0; nt < 2; ++nt) {
      v8f acc = {};
      acc = wmma16(a0, vf[nt][0], acc);
      acc = wmma16(a1, vf[nt][1], acc);
      store_tile_f16(oh, mt * 16, h * 32 + nt * 16, XSTRIDE, lane, acc, 1.0f, 0.0f);
    }
  }
  __syncthreads();

  // --- phase 3: output projection, K=256, + bp, f32 out ---
  const _Float16* wpT = wT + 3 * 65536;
  float* og = out + (size_t)win * NTOK * DIM;
  for (int t = 0; t < 2; ++t) {
    int col0 = (wave * 2 + t) << 4;
    float bb = bp[col0 + (lane & 15)];
    v16h bf[8];
#pragma unroll
    for (int ks = 0; ks < 8; ++ks) bf[ks] = load_fragB(wpT, col0, 256, ks * 32, lane);
    for (int mt = 0; mt < 4; ++mt) {
      v8f acc = {};
#pragma unroll
      for (int ks = 0; ks < 8; ++ks) {
        v16h a = load_fragA(oh, mt * 16, XSTRIDE, ks * 32, lane);
        acc = wmma16(a, bf[ks], acc);
      }
      int col = col0 + (lane & 15);
      int rb  = mt * 16 + ((lane >> 4) << 3);
#pragma unroll
      for (int r = 0; r < 8; ++r)
        og[(rb + r) * DIM + col] = acc[r] + bb;
    }
  }
}

extern "C" void kernel_launch(void* const* d_in, const int* in_sizes, int n_in,
                              void* d_out, int out_size, void* d_ws, size_t ws_size,
                              hipStream_t stream) {
  (void)in_sizes; (void)n_in; (void)out_size; (void)ws_size;
  const float* x     = (const float*)d_in[0];
  const float* wq    = (const float*)d_in[1];
  const float* bq    = (const float*)d_in[2];
  const float* wkv   = (const float*)d_in[3];
  const float* bkv   = (const float*)d_in[4];
  const float* table = (const float*)d_in[5];
  const float* wp    = (const float*)d_in[6];
  const float* bp    = (const float*)d_in[7];
  _Float16* wT = (_Float16*)d_ws;                    // 4 * 256*256 f16 = 512 KB

  hipFuncSetAttribute((const void*)wa_attn_kernel,
                      hipFuncAttributeMaxDynamicSharedMemorySize, SMEM_BYTES);

  wa_convert_weights<<<256, 256, 0, stream>>>(wq, wkv, wp, wT);
  wa_attn_kernel<<<NWIN, 256, SMEM_BYTES, stream>>>(x, bq, bkv, table, bp, wT,
                                                    (float*)d_out);
}
